// Seq2SeqDualModelAlignTimestep_72799695667784
// MI455X (gfx1250) — compile-verified, matrix operand order checked by reading, not run on hardware
//
#include <hip/hip_runtime.h>
#include <hip/hip_bf16.h>

typedef __attribute__((ext_vector_type(16))) _Float16 v16h;
typedef __attribute__((ext_vector_type(8)))  float    v8f;

#define DEVINL __device__ __forceinline__

// ---------------- model dims ----------------
constexpr int B_   = 32;
constexpr int L_   = 100;
constexpr int T_   = 50;
constexpr int V_   = 20000;
constexpr int H_DEC = 500, H_DIR = 250, H_ENC = 500;
// padded (WMMA-friendly) dims
constexpr int KENC = 768;   // 460+250=710 -> 768  (24 k-chunks of 32)
constexpr int NENC = 1008;  // 4*250=1000 -> 1008  (63 n-tiles)
constexpr int KDEC = 928;   // 400+500=900 -> 928  (29 k-chunks)
constexpr int NDEC = 2016;  // 4*500=2000 -> 2016  (126 n-tiles)
constexpr int KLIN = 512;   // 500 -> 512          (16 k-chunks)
constexpr float EPSA = 1e-3f;

// ---------------- workspace layout (bytes, all 256B-aligned) ----------------
constexpr size_t OFF_XHF    = 0;
constexpr size_t OFF_XHB    = OFF_XHF    + (size_t)L_*B_*KENC*2;
constexpr size_t OFF_INPZ   = OFF_XHB    + (size_t)L_*B_*KENC*2;
constexpr size_t OFF_WCAT   = OFF_INPZ   + (size_t)B_*L_*64*4;
constexpr size_t OFF_WDEC   = OFF_WCAT   + (size_t)2*NENC*KENC*2;
constexpr size_t OFF_WLIN   = OFF_WDEC   + (size_t)NDEC*KDEC*2;
constexpr size_t OFF_CENC   = OFF_WLIN   + (size_t)V_*KLIN*2;
constexpr size_t OFF_HS     = OFF_CENC   + (size_t)2*B_*256*4;
constexpr size_t OFF_ENCOUT = OFF_HS     + (size_t)2*L_*B_*256*4;
constexpr size_t OFF_HDEC   = OFF_ENCOUT + (size_t)B_*L_*512*4;
constexpr size_t OFF_CDEC   = OFF_HDEC   + (size_t)B_*512*4;
constexpr size_t OFF_ZDEC   = OFF_CDEC   + (size_t)B_*512*4;
constexpr size_t OFF_ZENC   = OFF_ZDEC   + (size_t)B_*NDEC*4;
constexpr size_t OFF_HWD    = OFF_ZENC   + (size_t)2*B_*NENC*4;
constexpr size_t OFF_HWZ    = OFF_HWD    + (size_t)B_*512*4;
constexpr size_t OFF_SD     = OFF_HWZ    + (size_t)B_*64*4;
constexpr size_t OFF_SZ     = OFF_SD     + (size_t)B_*L_*4;
constexpr size_t OFF_AROW   = OFF_SZ     + (size_t)B_*L_*4;
constexpr size_t OFF_CTX    = OFF_AROW   + (size_t)B_*L_*4;
constexpr size_t OFF_OUTS   = OFF_CTX    + (size_t)B_*512*4;
constexpr size_t OFF_OUTSH  = OFF_OUTS   + (size_t)B_*T_*H_ENC*4;
constexpr size_t OFF_ATTN   = OFF_OUTSH  + (size_t)B_*T_*512*2;
constexpr size_t OFF_LAM    = OFF_ATTN   + (size_t)B_*T_*L_*4;
constexpr size_t OFF_AMAXL  = OFF_LAM    + (size_t)B_*T_*4;
constexpr size_t OFF_M1     = OFF_AMAXL  + (size_t)B_*T_*4;
constexpr size_t OFF_LS1    = OFF_M1     + (size_t)B_*T_*4;
constexpr size_t OFF_M2     = OFF_LS1    + (size_t)B_*T_*4;
constexpr size_t OFF_LS2    = OFF_M2     + (size_t)B_*T_*4;
constexpr size_t OFF_AMAXV  = OFF_LS2    + (size_t)B_*T_*4;
constexpr size_t OFF_LOGITS = OFF_AMAXV  + (size_t)B_*T_*4;

// ---------------- output layout (f32 elements) ----------------
constexpr size_t O_PB = 0;
constexpr size_t O_HT = (size_t)B_*T_*V_;
constexpr size_t O_CT = O_HT + (size_t)B_*H_DEC;
constexpr size_t O_AP = O_CT + (size_t)B_*H_DEC;
constexpr size_t O_DP = O_AP + (size_t)B_*T_*300;

// ---------------- helpers ----------------
DEVINL float sigf(float x) { return 1.0f / (1.0f + __expf(-x)); }

DEVINL v8f wmma16(v16h a, v16h b, v8f c) {
  return __builtin_amdgcn_wmma_f32_16x16x32_f16(false, a, false, b, (short)0, c,
                                                false, false);
}

// A tile: row-major [M x lda] f16, tile at (m0, k0). 16-bit A layout:
// lanes 0-15 rows m0..m0+15 K={0..7,16..23}; lanes 16-31 same rows K+8.
DEVINL v16h load_a16(const _Float16* A, int lda, int m0, int k0, int lane) {
  int row  = m0 + (lane & 15);
  int kadd = (lane >> 4) << 3;
  const _Float16* p = A + (size_t)row * lda + k0 + kadd;
  v16h r;
#pragma unroll
  for (int j = 0; j < 16; ++j) { int k = (j < 8) ? j : (j + 8); r[j] = p[k]; }
  return r;
}

// B tile from weights stored row-major [N x ldw] (B[k][n] = W[n][k]).
DEVINL v16h load_b16(const _Float16* W, int ldw, int n0, int k0, int lane) {
  int col  = n0 + (lane & 15);
  int kadd = (lane >> 4) << 3;
  const _Float16* p = W + (size_t)col * ldw + k0 + kadd;
  v16h r;
#pragma unroll
  for (int j = 0; j < 16; ++j) { int k = (j < 8) ? j : (j + 8); r[j] = p[k]; }
  return r;
}

// C/D f32 16x16: lanes 0-15 N=lane M=m0+0..7 in regs 0..7; lanes 16-31 M+8.
DEVINL void store_c16(float* C, int ldc, int m0, int n0, int lane, v8f acc) {
  int n  = n0 + (lane & 15);
  int mb = m0 + ((lane >> 4) << 3);
#pragma unroll
  for (int r = 0; r < 8; ++r) C[(size_t)(mb + r) * ldc + n] = acc[r];
}

// ================= kernel 1: weight conversion to padded f16 =================
__global__ void k_prep(const float* wih_f, const float* whh_f,
                       const float* wih_b, const float* whh_b,
                       const float* dwih, const float* dwhh,
                       const float* wlin,
                       _Float16* wcat, _Float16* wdec, _Float16* wlinh) {
  constexpr int NW1 = 2 * NENC * KENC;
  constexpr int NW2 = NDEC * KDEC;
  constexpr int NW3 = V_ * KLIN;
  for (int i = blockIdx.x * blockDim.x + threadIdx.x; i < NW1 + NW2 + NW3;
       i += gridDim.x * blockDim.x) {
    if (i < NW1) {
      int dir = i / (NENC * KENC);
      int rem = i % (NENC * KENC);
      int n = rem / KENC, k = rem % KENC;
      float v = 0.0f;
      if (n < 1000) {
        if (k < 460)      v = (dir ? wih_b : wih_f)[n * 460 + k];
        else if (k < 710) v = (dir ? whh_b : whh_f)[n * 250 + (k - 460)];
      }
      wcat[i] = (_Float16)v;
    } else if (i < NW1 + NW2) {
      int rem = i - NW1;
      int n = rem / KDEC, k = rem % KDEC;
      float v = 0.0f;
      if (n < 2000) {
        if (k < 400)      v = dwih[n * 400 + k];
        else if (k < 900) v = dwhh[n * 500 + (k - 400)];
      }
      wdec[rem] = (_Float16)v;
    } else {
      int rem = i - NW1 - NW2;
      int n = rem / KLIN, k = rem % KLIN;
      wlinh[rem] = (_Float16)((k < 500) ? wlin[n * 500 + k] : 0.0f);
    }
  }
}

// ============ kernel 2: embeddings -> XH_f / XH_b (f16) + inp_z, c=0 ==========
__global__ void k_embed(const float* E_sent, const float* E_field,
                        const float* E_ppos, const float* E_pneg,
                        const int* value, const int* field,
                        const int* ppos, const int* pneg,
                        _Float16* xhf, _Float16* xhb, float* inpz,
                        float* cenc) {
  constexpr int NX = 2 * L_ * B_ * KENC;
  constexpr int NZ = B_ * L_ * 64;
  constexpr int NC = 2 * B_ * 256;
  for (int i = blockIdx.x * blockDim.x + threadIdx.x; i < NX + NZ + NC;
       i += gridDim.x * blockDim.x) {
    if (i < NX) {
      int half = i / (L_ * B_ * KENC);
      int rem  = i % (L_ * B_ * KENC);
      int t = rem / (B_ * KENC);
      int b = (rem / KENC) % B_;
      int col = rem % KENC;
      int ts = half ? (L_ - 1 - t) : t;
      int e = b * L_ + ts;
      float v = 0.0f;
      if      (col < 400) v = E_sent [value[e] * 400 + col];
      else if (col < 450) v = E_field[field[e] * 50 + (col - 400)];
      else if (col < 455) v = E_ppos [ppos [e] * 5  + (col - 450)];
      else if (col < 460) v = E_pneg [pneg [e] * 5  + (col - 455)];
      (half ? xhb : xhf)[rem] = (_Float16)v;
    } else if (i < NX + NZ) {
      int rem = i - NX;
      int b = rem / (L_ * 64), l = (rem / 64) % L_, j = rem % 64;
      int e = b * L_ + l;
      float v = 0.0f;
      if      (j < 50) v = E_field[field[e] * 50 + j];
      else if (j < 55) v = E_ppos [ppos [e] * 5 + (j - 50)];
      else if (j < 60) v = E_pneg [pneg [e] * 5 + (j - 55)];
      inpz[rem] = v;
    } else {
      cenc[i - NX - NZ] = 0.0f;
    }
  }
}

// ========= kernel 3: persistent bidirectional LSTM encoder (WMMA gates) ======
// A matrix (32 x 768 f16, 48KB) staged in LDS once per timestep; all 16 waves
// feed WMMA from LDS while the f16 weight stream stays on global_load_b128.
__global__ void k_encoder(_Float16* xhf, _Float16* xhb, const _Float16* wcat,
                          const float* b_f, const float* b_b,
                          float* zenc, float* cenc, float* hs) {
  __shared__ _Float16 As[B_ * KENC];            // 49152 bytes
  const int dir = blockIdx.x;
  _Float16* XH = dir ? xhb : xhf;
  const _Float16* W = wcat + (size_t)dir * NENC * KENC;
  const float* bias = dir ? b_b : b_f;
  float* Z  = zenc + (size_t)dir * B_ * NENC;
  float* C  = cenc + (size_t)dir * B_ * 256;
  float* HS = hs   + (size_t)dir * L_ * B_ * 256;
  const int lane = threadIdx.x & 31, wv = threadIdx.x >> 5;

  for (int t = 0; t < L_; ++t) {
    // stage A into LDS (global b128 -> ds_store)
    {
      const uint4* src = (const uint4*)(XH + (size_t)t * B_ * KENC);
      uint4* dst = (uint4*)As;
      for (int i = threadIdx.x; i < B_ * KENC / 8; i += blockDim.x)
        dst[i] = src[i];
    }
    __syncthreads();

    for (int tile = wv; tile < 2 * (NENC / 16); tile += 16) {
      int m0 = (tile / (NENC / 16)) * 16;
      int n0 = (tile % (NENC / 16)) * 16;
      v8f acc = {};
      for (int kt = 0; kt < KENC / 32; ++kt)
        acc = wmma16(load_a16(As, KENC, m0, kt * 32, lane),
                     load_b16(W, KENC, n0, kt * 32, lane), acc);
      store_c16(Z, NENC, m0, n0, lane, acc);
    }
    __threadfence(); __syncthreads();

    for (int idx = threadIdx.x; idx < B_ * H_DIR; idx += blockDim.x) {
      int b = idx / H_DIR, n = idx % H_DIR;
      const float* zb = Z + (size_t)b * NENC;
      float zi = zb[n]       + bias[n];
      float zf = zb[250 + n] + bias[250 + n];
      float zg = zb[500 + n] + bias[500 + n];
      float zo = zb[750 + n] + bias[750 + n];
      float c = sigf(zf) * C[b * 256 + n] + sigf(zi) * tanhf(zg);
      float h = sigf(zo) * tanhf(c);
      C[b * 256 + n] = c;
      HS[(size_t)t * B_ * 256 + b * 256 + n] = h;
      if (t < L_ - 1)
        XH[(size_t)(t + 1) * B_ * KENC + b * KENC + 460 + n] = (_Float16)h;
    }
    __threadfence(); __syncthreads();
  }
}

// ===== kernel 4: enc_out assembly + torch-style .view for h0_dec / c0_dec ====
__global__ void k_encout_h0(const float* hs, const float* cenc,
                            float* encout, float* hdec, float* cdec) {
  constexpr int NE = B_ * L_ * 512;
  constexpr int NH = B_ * 512;
  const float* HSF = hs;
  const float* HSB = hs + (size_t)L_ * B_ * 256;
  for (int i = blockIdx.x * blockDim.x + threadIdx.x; i < NE + NH;
       i += gridDim.x * blockDim.x) {
    if (i < NE) {
      int b = i / (L_ * 512), l = (i / 512) % L_, j = i % 512;
      float v = 0.0f;
      if (j < 250)      v = HSF[(size_t)l * B_ * 256 + b * 256 + j];
      else if (j < 500) v = HSB[(size_t)(L_ - 1 - l) * B_ * 256 + b * 256 + (j - 250)];
      encout[i] = v;
    } else {
      int rem = i - NE;
      int b = rem / 512, j = rem % 512;
      float hv = 0.0f, cv = 0.0f;
      if (j < 500) {
        int flat = b * 500 + j;
        int d = flat / (B_ * 250);
        int r = flat % (B_ * 250);
        int b2 = r / 250, k = r % 250;
        hv = hs[(size_t)d * L_ * B_ * 256 + (size_t)(L_ - 1) * B_ * 256 + b2 * 256 + k];
        cv = cenc[(size_t)d * B_ * 256 + b2 * 256 + k];
      }
      hdec[rem] = hv;
      cdec[rem] = cv;
    }
  }
}

// ========== kernel 5: persistent decoder (WMMA gates + dual attention) =======
// [x_t | h] (32 x 928 f16, 58KB) is built directly in LDS each step.
__global__ void k_decoder(const float* E_sent, const int* sent,
                          const _Float16* wdec, const float* dec_b,
                          const float* W_d, const float* W_z,
                          const float* W_out, const float* b_out,
                          const float* w_l, const float* b_l,
                          const float* encout, const float* inpz,
                          float* Z, float* h, float* c,
                          float* hwd, float* hwz, float* sd, float* sz,
                          float* arow, float* ctx,
                          float* outs, _Float16* outsh, float* attn,
                          float* lam, int* amaxl, float* out) {
  __shared__ _Float16 xs[B_ * KDEC];            // 59392 bytes
  const int lane = threadIdx.x & 31, wv = threadIdx.x >> 5;
  for (int t = 0; t < T_; ++t) {
    // phase 1: xs = [E_sent[sent[:,t]] | h | 0pad] in f16 (straight into LDS)
    for (int i = threadIdx.x; i < B_ * KDEC; i += blockDim.x) {
      int b = i / KDEC, col = i % KDEC;
      float v = 0.0f;
      if (col < 400)      v = E_sent[sent[b * T_ + t] * 400 + col];
      else if (col < 900) v = h[b * 512 + (col - 400)];
      xs[i] = (_Float16)v;
    }
    __syncthreads();

    // phase 2: Z[32,2016] = xs @ Wdec^T  (WMMA, A from LDS)
    for (int tile = wv; tile < 2 * (NDEC / 16); tile += 16) {
      int m0 = (tile / (NDEC / 16)) * 16;
      int n0 = (tile % (NDEC / 16)) * 16;
      v8f acc = {};
      for (int kt = 0; kt < KDEC / 32; ++kt)
        acc = wmma16(load_a16(xs, KDEC, m0, kt * 32, lane),
                     load_b16(wdec, KDEC, n0, kt * 32, lane), acc);
      store_c16(Z, NDEC, m0, n0, lane, acc);
    }
    __threadfence(); __syncthreads();

    // phase 3: LSTM cell
    for (int i = threadIdx.x; i < B_ * H_DEC; i += blockDim.x) {
      int b = i / H_DEC, n = i % H_DEC;
      const float* zb = Z + (size_t)b * NDEC;
      float zi = zb[n]        + dec_b[n];
      float zf = zb[500 + n]  + dec_b[500 + n];
      float zg = zb[1000 + n] + dec_b[1000 + n];
      float zo = zb[1500 + n] + dec_b[1500 + n];
      float cc = sigf(zf) * c[b * 512 + n] + sigf(zi) * tanhf(zg);
      float hh = sigf(zo) * tanhf(cc);
      c[b * 512 + n] = cc;
      h[b * 512 + n] = hh;
    }
    __threadfence(); __syncthreads();

    // phase 4: hWd = h @ W_d, hWz = h @ W_z
    for (int i = threadIdx.x; i < B_ * 500; i += blockDim.x) {
      int b = i / 500, j = i % 500;
      float s = 0.0f;
      for (int k = 0; k < 500; ++k) s += h[b * 512 + k] * W_d[k * 500 + j];
      hwd[b * 512 + j] = s;
    }
    for (int i = threadIdx.x; i < B_ * 60; i += blockDim.x) {
      int b = i / 60, j = i % 60;
      float s = 0.0f;
      for (int k = 0; k < 500; ++k) s += h[b * 512 + k] * W_z[k * 60 + j];
      hwz[b * 64 + j] = s;
    }
    __threadfence(); __syncthreads();

    // phase 5: scores s_d, s_z
    for (int i = threadIdx.x; i < B_ * L_; i += blockDim.x) {
      int b = i / L_, l = i % L_;
      float s1 = 0.0f;
      const float* eo = encout + (size_t)(b * L_ + l) * 512;
      for (int j = 0; j < 500; ++j) s1 += hwd[b * 512 + j] * eo[j];
      sd[i] = s1;
      float s2 = 0.0f;
      const float* iz = inpz + (size_t)(b * L_ + l) * 64;
      for (int j = 0; j < 60; ++j) s2 += hwz[b * 64 + j] * iz[j];
      sz[i] = s2;
    }
    __threadfence(); __syncthreads();

    // phase 6: dual softmax product, normalize w/ EPS, argmax  (per-b lane)
    if (threadIdx.x < B_) {
      int b = threadIdx.x;
      float m1 = -1e30f, m2 = -1e30f;
      for (int l = 0; l < L_; ++l) {
        m1 = fmaxf(m1, sd[b * L_ + l]);
        m2 = fmaxf(m2, sz[b * L_ + l]);
      }
      float d1 = 0.0f, d2 = 0.0f;
      for (int l = 0; l < L_; ++l) {
        d1 += __expf(sd[b * L_ + l] - m1);
        d2 += __expf(sz[b * L_ + l] - m2);
      }
      float asum = 0.0f;
      for (int l = 0; l < L_; ++l) {
        float a = (__expf(sd[b * L_ + l] - m1) / d1) *
                  (__expf(sz[b * L_ + l] - m2) / d2);
        arow[b * L_ + l] = a;
        asum += a;
      }
      float inv = 1.0f / (asum + EPSA);
      float bm = -1.0f; int bi = 0;
      for (int l = 0; l < L_; ++l) {
        float a = arow[b * L_ + l] * inv;
        arow[b * L_ + l] = a;
        attn[(size_t)(b * T_ + t) * L_ + l] = a;
        if (a > bm) { bm = a; bi = l; }
      }
      amaxl[b * T_ + t] = bi;
    }
    __threadfence(); __syncthreads();

    // phase 7: ctx = a @ enc_out
    for (int i = threadIdx.x; i < B_ * 500; i += blockDim.x) {
      int b = i / 500, j = i % 500;
      float s = 0.0f;
      for (int l = 0; l < L_; ++l)
        s += arow[b * L_ + l] * encout[(size_t)(b * L_ + l) * 512 + j];
      ctx[b * 512 + j] = s;
    }
    __threadfence(); __syncthreads();

    // phase 8: out_t = tanh([h|ctx] @ W_out + b_out)
    for (int i = threadIdx.x; i < B_ * 512; i += blockDim.x) {
      int b = i / 512, j = i % 512;
      if (j < 500) {
        float s = b_out[j];
        for (int k = 0; k < 500; ++k) s += h[b * 512 + k] * W_out[k * 500 + j];
        for (int k = 0; k < 500; ++k) s += ctx[b * 512 + k] * W_out[(500 + k) * 500 + j];
        float o = tanhf(s);
        outs[(size_t)(b * T_ + t) * H_ENC + j] = o;
        outsh[(size_t)(b * T_ + t) * 512 + j] = (_Float16)o;
      } else {
        outsh[(size_t)(b * T_ + t) * 512 + j] = (_Float16)0.0f;
      }
    }

    // phase 9: lambda gate
    if (threadIdx.x < B_) {
      int b = threadIdx.x;
      float s = b_l[0];
      for (int k = 0; k < 500; ++k) s += h[b * 512 + k] * w_l[k];
      for (int k = 0; k < 500; ++k) s += ctx[b * 512 + k] * w_l[500 + k];
      const float* xt = E_sent + (size_t)sent[b * T_ + t] * 400;
      for (int k = 0; k < 400; ++k) s += xt[k] * w_l[1000 + k];
      lam[b * T_ + t] = sigf(s);
    }

    // phase 10: final state out
    if (t == T_ - 1) {
      for (int i = threadIdx.x; i < B_ * H_DEC; i += blockDim.x) {
        int b = i / H_DEC, j = i % H_DEC;
        out[O_HT + i] = h[b * 512 + j];
        out[O_CT + i] = c[b * 512 + j];
      }
    }
    __threadfence(); __syncthreads();
  }
}

// ============== kernel 6: vocab projection GEMM (WMMA, the big one) ==========
// Block = one m-tile (16 rows) x 8 n-tiles. A slab (16x512 f16) staged in LDS
// and shared by all 8 waves; weight stream stays on global with prefetch.
constexpr int NGRP = (V_ / 16 + 7) / 8;     // 157 n-groups of 8 tiles
__global__ void k_vocab(const _Float16* outsh, const _Float16* wlinh,
                        const float* b_lin, float* logits) {
  __shared__ _Float16 As[16 * KLIN];          // 16KB
  const int lane = threadIdx.x & 31, wv = threadIdx.x >> 5;
  const int mt = blockIdx.x / NGRP;
  const int ng = blockIdx.x % NGRP;
  const int m0 = mt * 16;

  {
    const uint4* src = (const uint4*)(outsh + (size_t)m0 * KLIN);
    uint4* dst = (uint4*)As;
    for (int i = threadIdx.x; i < 16 * KLIN / 8; i += blockDim.x)
      dst[i] = src[i];
  }
  __syncthreads();

  const int nt = ng * 8 + wv;
  if (nt >= V_ / 16) return;
  const int n0 = nt * 16;

  __builtin_prefetch(wlinh + (size_t)n0 * KLIN, 0, 3);
  v8f acc = {};
  for (int kt = 0; kt < KLIN / 32; ++kt) {
    __builtin_prefetch(wlinh + (size_t)n0 * KLIN + (kt + 1) * 32, 0, 3);
    acc = wmma16(load_a16(As, KLIN, 0, kt * 32, lane),
                 load_b16(wlinh, KLIN, n0, kt * 32, lane), acc);
  }
  const int n  = n0 + (lane & 15);
  const int mb = m0 + ((lane >> 4) << 3);
  const float bl = b_lin[n];
#pragma unroll
  for (int r = 0; r < 8; ++r)
    logits[(size_t)(mb + r) * V_ + n] = acc[r] + bl;
}

// ======= kernel 7: per-row max / log-sum-exp of model logits (p_mod) =========
__global__ void k_rowlse(const float* logits, float* m1, float* ls1) {
  __shared__ float red[256];
  const int row = blockIdx.x;
  const float* p = logits + (size_t)row * V_;
  float m = -1e30f;
  for (int v = threadIdx.x; v < V_; v += blockDim.x) m = fmaxf(m, p[v]);
  red[threadIdx.x] = m;
  __syncthreads();
  for (int s = 128; s > 0; s >>= 1) {
    if (threadIdx.x < s) red[threadIdx.x] = fmaxf(red[threadIdx.x], red[threadIdx.x + s]);
    __syncthreads();
  }
  m = red[0];
  __syncthreads();
  float acc = 0.0f;
  for (int v = threadIdx.x; v < V_; v += blockDim.x) acc += __expf(p[v] - m);
  red[threadIdx.x] = acc;
  __syncthreads();
  for (int s = 128; s > 0; s >>= 1) {
    if (threadIdx.x < s) red[threadIdx.x] += red[threadIdx.x + s];
    __syncthreads();
  }
  if (threadIdx.x == 0) { m1[row] = m; ls1[row] = logf(red[0]); }
}

// == kernel 8: y = lam*p_lex + (1-lam)*p_mod in-place; row max/argmax/lse =====
__global__ void k_combine(float* logits, const float* attn,
                          const float* align, const float* lam,
                          const float* m1, const float* ls1,
                          float* m2, float* ls2, int* amaxv) {
  __shared__ float red[256];
  __shared__ int   redi[256];
  const int row = blockIdx.x;            // b*T + t
  const int b = row / T_;
  const float lm = lam[row];
  const float mm = m1[row], ll = ls1[row];
  float* y = logits + (size_t)row * V_;
  const float* ar = attn + (size_t)row * L_;
  float bm = -1e30f; int bi = 0;
  for (int v = threadIdx.x; v < V_; v += blockDim.x) {
    float pl = 0.0f;
    const float* al = align + (size_t)b * L_ * V_ + v;
    for (int l = 0; l < L_; ++l) pl += ar[l] * al[(size_t)l * V_];
    float pm = y[v] - mm - ll;               // log-softmaxed model dist
    float yy = lm * pl + (1.0f - lm) * pm;
    y[v] = yy;
    if (yy > bm) { bm = yy; bi = v; }
  }
  red[threadIdx.x] = bm; redi[threadIdx.x] = bi;
  __syncthreads();
  for (int s = 128; s > 0; s >>= 1) {
    if (threadIdx.x < s && red[threadIdx.x + s] > red[threadIdx.x]) {
      red[threadIdx.x] = red[threadIdx.x + s];
      redi[threadIdx.x] = redi[threadIdx.x + s];
    }
    __syncthreads();
  }
  const float rm = red[0];
  if (threadIdx.x == 0) { m2[row] = rm; amaxv[row] = redi[0]; }
  __syncthreads();
  float acc = 0.0f;
  for (int v = threadIdx.x; v < V_; v += blockDim.x) acc += __expf(y[v] - rm);
  red[threadIdx.x] = acc;
  __syncthreads();
  for (int s = 128; s > 0; s >>= 1) {
    if (threadIdx.x < s) red[threadIdx.x] += red[threadIdx.x + s];
    __syncthreads();
  }
  if (threadIdx.x == 0) ls2[row] = logf(red[0]);
}

// ============ kernel 9: p_bias = y - m2 - ls2  ->  d_out ====================
__global__ void k_final(const float* logits, const float* m2, const float* ls2,
                        float* out) {
  for (size_t i = (size_t)blockIdx.x * blockDim.x + threadIdx.x;
       i < (size_t)B_ * T_ * V_; i += (size_t)gridDim.x * blockDim.x) {
    size_t row = i / V_;
    out[O_PB + i] = logits[i] - m2[row] - ls2[row];
  }
}

// ============ kernel 10: E_target gathers (attn_pred, decoder_pred) =========
__global__ void k_gather(const float* E_target, const int* value,
                         const int* amaxl, const int* amaxv, float* out) {
  constexpr int N = B_ * T_ * 300;
  for (int i = blockIdx.x * blockDim.x + threadIdx.x; i < 2 * N;
       i += gridDim.x * blockDim.x) {
    if (i < N) {
      int row = i / 300, k = i % 300;
      int b = row / T_;
      int w = value[b * L_ + amaxl[row]];
      out[O_AP + i] = E_target[(size_t)w * 300 + k];
    } else {
      int j = i - N;
      int row = j / 300, k = j % 300;
      out[O_DP + j] = E_target[(size_t)amaxv[row] * 300 + k];
    }
  }
}

// ============================== launch ======================================
extern "C" void kernel_launch(void* const* d_in, const int* in_sizes, int n_in,
                              void* d_out, int out_size, void* d_ws,
                              size_t ws_size, hipStream_t stream) {
  const float* align_prob = (const float*)d_in[1];
  const float* E_sent     = (const float*)d_in[2];
  const float* E_field    = (const float*)d_in[3];
  const float* E_ppos     = (const float*)d_in[4];
  const float* E_pneg     = (const float*)d_in[5];
  const float* E_target   = (const float*)d_in[6];
  const float* enc_Wih_f  = (const float*)d_in[7];
  const float* enc_Whh_f  = (const float*)d_in[8];
  const float* enc_b_f    = (const float*)d_in[9];
  const float* enc_Wih_b  = (const float*)d_in[10];
  const float* enc_Whh_b  = (const float*)d_in[11];
  const float* enc_b_b    = (const float*)d_in[12];
  const float* dec_Wih    = (const float*)d_in[13];
  const float* dec_Whh    = (const float*)d_in[14];
  const float* dec_b      = (const float*)d_in[15];
  const float* W_d        = (const float*)d_in[16];
  const float* W_z        = (const float*)d_in[17];
  const float* W_out      = (const float*)d_in[18];
  const float* b_out      = (const float*)d_in[19];
  const float* w_l        = (const float*)d_in[20];
  const float* b_l        = (const float*)d_in[21];
  const float* W_lin      = (const float*)d_in[22];
  const float* b_lin      = (const float*)d_in[23];
  const int*   sent       = (const int*)d_in[24];
  const int*   value      = (const int*)d_in[25];
  const int*   field      = (const int*)d_in[26];
  const int*   ppos       = (const int*)d_in[27];
  const int*   pneg       = (const int*)d_in[28];

  char* ws = (char*)d_ws;
  _Float16* xhf   = (_Float16*)(ws + OFF_XHF);
  _Float16* xhb   = (_Float16*)(ws + OFF_XHB);
  float*    inpz  = (float*)   (ws + OFF_INPZ);
  _Float16* wcat  = (_Float16*)(ws + OFF_WCAT);
  _Float16* wdec  = (_Float16*)(ws + OFF_WDEC);
  _Float16* wlinh = (_Float16*)(ws + OFF_WLIN);
  float*    cenc  = (float*)   (ws + OFF_CENC);
  float*    hs    = (float*)   (ws + OFF_HS);
  float*    encout= (float*)   (ws + OFF_ENCOUT);
  float*    hdec  = (float*)   (ws + OFF_HDEC);
  float*    cdec  = (float*)   (ws + OFF_CDEC);
  float*    zdec  = (float*)   (ws + OFF_ZDEC);
  float*    zenc  = (float*)   (ws + OFF_ZENC);
  float*    hwd   = (float*)   (ws + OFF_HWD);
  float*    hwz   = (float*)   (ws + OFF_HWZ);
  float*    sd    = (float*)   (ws + OFF_SD);
  float*    sz    = (float*)   (ws + OFF_SZ);
  float*    arow  = (float*)   (ws + OFF_AROW);
  float*    ctx   = (float*)   (ws + OFF_CTX);
  float*    outs  = (float*)   (ws + OFF_OUTS);
  _Float16* outsh = (_Float16*)(ws + OFF_OUTSH);
  float*    attn  = (float*)   (ws + OFF_ATTN);
  float*    lam   = (float*)   (ws + OFF_LAM);
  int*      amaxl = (int*)     (ws + OFF_AMAXL);
  float*    m1    = (float*)   (ws + OFF_M1);
  float*    ls1   = (float*)   (ws + OFF_LS1);
  float*    m2    = (float*)   (ws + OFF_M2);
  float*    ls2   = (float*)   (ws + OFF_LS2);
  int*      amaxv = (int*)     (ws + OFF_AMAXV);
  float*    logits= (float*)   (ws + OFF_LOGITS);
  float*    out   = (float*)d_out;

  k_prep<<<8192, 256, 0, stream>>>(enc_Wih_f, enc_Whh_f, enc_Wih_b, enc_Whh_b,
                                   dec_Wih, dec_Whh, W_lin, wcat, wdec, wlinh);
  k_embed<<<8192, 256, 0, stream>>>(E_sent, E_field, E_ppos, E_pneg,
                                    value, field, ppos, pneg,
                                    xhf, xhb, inpz, cenc);
  k_encoder<<<2, 512, 0, stream>>>(xhf, xhb, wcat, enc_b_f, enc_b_b,
                                   zenc, cenc, hs);
  k_encout_h0<<<4096, 256, 0, stream>>>(hs, cenc, encout, hdec, cdec);
  k_decoder<<<1, 512, 0, stream>>>(E_sent, sent, wdec, dec_b, W_d, W_z,
                                   W_out, b_out, w_l, b_l, encout, inpz,
                                   zdec, hdec, cdec, hwd, hwz, sd, sz,
                                   arow, ctx, outs, outsh, attn, lam, amaxl,
                                   out);
  k_vocab<<<100 * NGRP, 256, 0, stream>>>(outsh, wlinh, b_lin, logits);
  k_rowlse<<<B_ * T_, 256, 0, stream>>>(logits, m1, ls1);
  k_combine<<<B_ * T_, 256, 0, stream>>>(logits, attn, align_prob, lam,
                                         m1, ls1, m2, ls2, amaxv);
  k_final<<<16384, 256, 0, stream>>>(logits, m2, ls2, out);
  k_gather<<<2048, 256, 0, stream>>>(E_target, value, amaxl, amaxv, out);
}
